// NewGAT_16080357556441
// MI455X (gfx1250) — matrix operand order repaired and checked by reading, not compiled, not run on hardware
//
#include <hip/hip_runtime.h>

// ---------------------------------------------------------------------------
// GAT layer for MI455X (gfx1250, wave32, WMMA).
// N=4096, C=512, F=64, H=8, E=131072.
// All fp32->bf16 conversion + WMMA fragment swizzling is hoisted into
// one-shot layout kernels; both GEMM loops are pure b128 loads + v_wmma.
// ---------------------------------------------------------------------------

#define GAT_N 4096
#define GAT_C 512
#define GAT_F 64
#define GAT_H 8
#define GAT_E 131072
#define NTILES (GAT_N / 16)          // 256 sixteen-row tiles
#define KTILES (GAT_C / 32)          // 16 K-tiles for the projection
#define JTILES (GAT_N / 32)          // 128 thirty-two-row key tiles
#define ADJ_WORDS (GAT_N / 32)       // 128 u32 per row
#define NEG_BIG (-9.0e15f)
#define GAT_ALPHA 0.2f

typedef __attribute__((ext_vector_type(16))) __bf16 v16bf;
typedef __attribute__((ext_vector_type(8)))  float  v8f;

// 32-byte fragment viewed as u32s / uint4s / bf16x16 (little-endian:
// u[i] low half = element 2i, high half = element 2i+1).
union V16 {
    unsigned u[8];
    uint4    q[2];
    v16bf    v;
};

static __device__ __forceinline__ unsigned short f2bfbits(float f) {
    unsigned u = __float_as_uint(f);
    u += 0x7FFFu + ((u >> 16) & 1u);        // round-to-nearest-even
    return (unsigned short)(u >> 16);
}
static __device__ __forceinline__ unsigned packbf(float lo, float hi) {
    return (unsigned)f2bfbits(lo) | ((unsigned)f2bfbits(hi) << 16);
}

// Fragment element indices (16 bf16 = 32 bytes per lane slot).
static __device__ __forceinline__ size_t xa_idx(int it, int kk, int lane) {
    return (((size_t)it * KTILES + kk) * 32 + lane) * 16;
}
static __device__ __forceinline__ size_t wf_idx(int hd, int kk, int ft, int lane) {
    return ((((size_t)hd * KTILES + kk) * 4 + ft) * 32 + lane) * 16;
}
static __device__ __forceinline__ size_t hb_idx(int hd, int jt, int ft, int lane) {
    return ((((size_t)hd * JTILES + jt) * 4 + ft) * 32 + lane) * 16;
}

// ---------------------------------------------------------------------------
// Kernel 0a: x (f32, row-major) -> A-fragment-layout bf16.
// One wave per (n-tile, k-tile). Lane's 16 elements are one contiguous
// 16-float run: [kb+hi16, +16).
// ---------------------------------------------------------------------------
__global__ __launch_bounds__(32)
void k_convx(const float* __restrict__ x, unsigned short* __restrict__ xa) {
    const int it   = blockIdx.x >> 4;
    const int kk   = blockIdx.x & 15;
    const int lane = threadIdx.x;
    const int lo   = lane & 15;
    const bool hiL = lane >= 16;
    const int row  = it * 16 + lo;

    const float* src = x + (size_t)row * GAT_C + kk * 32 + (hiL ? 16 : 0);
    float4 r0 = *(const float4*)(src);
    float4 r1 = *(const float4*)(src + 4);
    float4 r2 = *(const float4*)(src + 8);
    float4 r3 = *(const float4*)(src + 12);

    V16 f;
    f.u[0] = packbf(r0.x, r0.y); f.u[1] = packbf(r0.z, r0.w);
    f.u[2] = packbf(r1.x, r1.y); f.u[3] = packbf(r1.z, r1.w);
    f.u[4] = packbf(r2.x, r2.y); f.u[5] = packbf(r2.z, r2.w);
    f.u[6] = packbf(r3.x, r3.y); f.u[7] = packbf(r3.z, r3.w);

    uint4* dst = (uint4*)(xa + xa_idx(it, kk, lane));
    dst[0] = f.q[0];
    dst[1] = f.q[1];
}

// ---------------------------------------------------------------------------
// Kernel 0b: W (f32, (H,C,F) row-major) -> B-fragment-layout bf16.
// One wave per (head, k-tile, f-tile); elem e <-> row kb+(lane>=16?16:0)+e,
// col ft*16+(lane&15)  (stride-64 gather, one-time cost).
// ---------------------------------------------------------------------------
__global__ __launch_bounds__(32)
void k_convw(const float* __restrict__ W, unsigned short* __restrict__ wf) {
    const int hd   = blockIdx.x >> 6;
    const int kk   = (blockIdx.x >> 2) & 15;
    const int ft   = blockIdx.x & 3;
    const int lane = threadIdx.x;
    const int lo   = lane & 15;
    const bool hiL = lane >= 16;

    const float* src =
        W + ((size_t)hd * GAT_C + kk * 32 + (hiL ? 16 : 0)) * GAT_F + ft * 16 + lo;
    V16 f;
#pragma unroll
    for (int i = 0; i < 8; ++i)
        f.u[i] = packbf(src[(2 * i) * GAT_F], src[(2 * i + 1) * GAT_F]);

    uint4* dst = (uint4*)(wf + wf_idx(hd, kk, ft, lane));
    dst[0] = f.q[0];
    dst[1] = f.q[1];
}

// ---------------------------------------------------------------------------
// Kernel 1: h = x @ W[head] via pure fragment loads + WMMA.  Stores h in
// B-fragment layout (for the attention PV matmul) and computes s_src/s_dst.
// One wave per (head, 16-row n-tile).
// ---------------------------------------------------------------------------
__global__ __launch_bounds__(32)
void k_project(const unsigned short* __restrict__ xa,
               const unsigned short* __restrict__ wf,
               const float* __restrict__ a, unsigned short* __restrict__ hb,
               float* __restrict__ ssrc, float* __restrict__ sdst) {
    const int wg   = blockIdx.x;
    const int hd   = wg >> 8;            // head
    const int it   = wg & 255;           // n-tile
    const int lane = threadIdx.x;
    const int lo   = lane & 15;
    const bool hiL = lane >= 16;

    __shared__ float htile[16][GAT_F + 1];

    v8f acc[4] = {};                     // 4 f-tiles of 16 cols
    for (int kk = 0; kk < KTILES; ++kk) {
        V16 af;
        const uint4* asrc = (const uint4*)(xa + xa_idx(it, kk, lane));
        af.q[0] = asrc[0];
        af.q[1] = asrc[1];
#pragma unroll
        for (int ft = 0; ft < 4; ++ft) {
            V16 bf;
            const uint4* bsrc = (const uint4*)(wf + wf_idx(hd, kk, ft, lane));
            bf.q[0] = bsrc[0];
            bf.q[1] = bsrc[1];
            acc[ft] = __builtin_amdgcn_wmma_f32_16x16x32_bf16(
                false, af.v, false, bf.v, (short)0, acc[ft], false, false);
        }
    }

    // Store h in B-fragment layout: this lane's 8 C-rows at column ft*16+lo
    // are 8 contiguous fragment elements -> one packed uint4 store per ft.
    const int jt     = it >> 1;                       // 32-row key tile
    const int lane_p = lo + ((it & 1) ? 16 : 0);      // fragment lane slot
    const int ebase  = hiL ? 8 : 0;                   // fragment elem offset
#pragma unroll
    for (int ft = 0; ft < 4; ++ft) {
        uint4 pk;
        pk.x = packbf(acc[ft][0], acc[ft][1]);
        pk.y = packbf(acc[ft][2], acc[ft][3]);
        pk.z = packbf(acc[ft][4], acc[ft][5]);
        pk.w = packbf(acc[ft][6], acc[ft][7]);
        *(uint4*)(hb + hb_idx(hd, jt, ft, lane_p) + ebase) = pk;
        // f32 copy into LDS for the s_src/s_dst dot products.
#pragma unroll
        for (int e2 = 0; e2 < 8; ++e2)
            htile[e2 + (hiL ? 8 : 0)][ft * 16 + lo] = acc[ft][e2];
    }
    __syncthreads();

    if (lane < 16) {
        const float* asrc = a + hd * 2 * GAT_F;
        const float* adst = asrc + GAT_F;
        float s1 = 0.f, s2 = 0.f;
#pragma unroll
        for (int f = 0; f < GAT_F; ++f) {
            float hv = htile[lane][f];
            s1 += hv * asrc[f];
            s2 += hv * adst[f];
        }
        ssrc[hd * GAT_N + it * 16 + lane] = s1;
        sdst[hd * GAT_N + it * 16 + lane] = s2;
    }
}

// ---------------------------------------------------------------------------
// Kernel 2a: adjacency bitmask init + self loops.
// ---------------------------------------------------------------------------
__global__ __launch_bounds__(128)
void k_adj_init(unsigned* __restrict__ adj) {
    const int row = blockIdx.x;
    const int t   = threadIdx.x;         // 0..127
    adj[(size_t)row * ADJ_WORDS + t] =
        (t == (row >> 5)) ? (1u << (row & 31)) : 0u;
}

// Kernel 2b: scatter edges: adj[i, j] = 1.
__global__ __launch_bounds__(256)
void k_adj_scatter(const long long* __restrict__ e0,
                   const long long* __restrict__ e1,
                   unsigned* __restrict__ adj) {
    const int e = blockIdx.x * 256 + threadIdx.x;
    if (e < GAT_E) {
        int i = (int)e0[e];
        int j = (int)e1[e];
        atomicOr(&adj[(size_t)i * ADJ_WORDS + (j >> 5)], 1u << (j & 31));
    }
}

// ---------------------------------------------------------------------------
// Kernel 3: flash-softmax attention. One wave per (head, 16-row i-tile).
// j-tiles of 32 keys; P (16x32 bf16) @ V (32x64 bf16) via 4 WMMAs/tile.
// V fragments come straight from the pre-swizzled hb buffer (2 x b128/lane).
// ---------------------------------------------------------------------------
__global__ __launch_bounds__(32)
void k_attn(const unsigned short* __restrict__ hb,
            const float* __restrict__ ssrc, const float* __restrict__ sdst,
            const unsigned* __restrict__ adj, float* __restrict__ outp) {
    const int wg   = blockIdx.x;
    const int hd   = wg >> 8;
    const int it   = wg & 255;
    const int lane = threadIdx.x;
    const int lo   = lane & 15;
    const bool hiL = lane >= 16;
    const int arow = it * 16 + lo;

    __shared__ float sm[16], sl[16], stmax[16], stsum[16], sds[32];

    const float ssr = ssrc[hd * GAT_N + arow];
    if (lane < 16) { sm[lane] = NEG_BIG; sl[lane] = 0.f; }
    v8f acc[4] = {};
    __syncthreads();

    for (int jb = 0; jb < JTILES; ++jb) {
        // Stage the 32 s_dst values for this key tile (coalesced).
        sds[lane] = sdst[hd * GAT_N + jb * 32 + lane];
        if (jb + 1 < JTILES)                          // global_prefetch_b8
            __builtin_prefetch(hb + hb_idx(hd, jb + 1, 0, lane), 0, 3);

        const unsigned aw = adj[(size_t)arow * ADJ_WORDS + jb];
        __syncthreads();

        // Scores for my A-row (16 of the 32 j's per lane, contiguous j = 16h + e order).
        float ev[16];
        float tm = NEG_BIG;
#pragma unroll
        for (int e = 0; e < 16; ++e) {
            int kk = e + (hiL ? 16 : 0);                      // j - jbase
            float v = ssr + sds[kk];
            v = v > 0.f ? v : GAT_ALPHA * v;                  // leaky relu
            if (!((aw >> kk) & 1u)) v = NEG_BIG;              // mask
            ev[e] = v;
            tm = fmaxf(tm, v);
        }
        tm = fmaxf(tm, __shfl_xor(tm, 16, 32));               // combine half-rows
        if (lane < 16) stmax[lane] = tm;
        __syncthreads();

        // Probabilities for my row with updated max (packed bf16 A frag).
        const float mn_a = fmaxf(sm[lo], tm);
        V16 pf;
        float ts = 0.f;
#pragma unroll
        for (int i = 0; i < 8; ++i) {
            float p0 = __expf(ev[2 * i]     - mn_a);
            float p1 = __expf(ev[2 * i + 1] - mn_a);
            ts += p0 + p1;
            pf.u[i] = packbf(p0, p1);
        }
        ts += __shfl_xor(ts, 16, 32);
        if (lane < 16) stsum[lane] = ts;

        // Rescale accumulators (C-layout rows) with old running max.
#pragma unroll
        for (int e2 = 0; e2 < 8; ++e2) {
            int r    = e2 + (hiL ? 8 : 0);
            float mo = sm[r];
            float sc = __expf(mo - fmaxf(mo, stmax[r]));
#pragma unroll
            for (int ft = 0; ft < 4; ++ft) acc[ft][e2] *= sc;
        }
        __syncthreads();   // all lanes done reading old sm
        if (lane < 16) {
            float mo = sm[lane];
            float mn = fmaxf(mo, stmax[lane]);
            sl[lane] = sl[lane] * __expf(mo - mn) + stsum[lane];
            sm[lane] = mn;
        }

        // P @ V : 4 WMMAs; B fragments are contiguous 32-byte loads.
#pragma unroll
        for (int ft = 0; ft < 4; ++ft) {
            V16 bf;
            const uint4* src = (const uint4*)(hb + hb_idx(hd, jb, ft, lane));
            bf.q[0] = src[0];
            bf.q[1] = src[1];
            acc[ft] = __builtin_amdgcn_wmma_f32_16x16x32_bf16(
                false, pf.v, false, bf.v, (short)0, acc[ft], false, false);
        }
        __syncthreads();   // sm/sds re-read next iteration
    }

    // Epilogue: normalize and store pre-softmax output (n, h*F+f).
#pragma unroll
    for (int e2 = 0; e2 < 8; ++e2) {
        int r     = e2 + (hiL ? 8 : 0);
        float inv = 1.0f / sl[r];
        int row   = it * 16 + r;
#pragma unroll
        for (int ft = 0; ft < 4; ++ft)
            outp[(size_t)row * (GAT_H * GAT_F) + hd * GAT_F + ft * 16 + lo] =
                acc[ft][e2] * inv;
    }
}

// ---------------------------------------------------------------------------
// Kernel 4: final row softmax over 512 columns.
// ---------------------------------------------------------------------------
__global__ __launch_bounds__(256)
void k_rowsoftmax(const float* __restrict__ outp, float* __restrict__ out) {
    const int row = blockIdx.x;
    const int t   = threadIdx.x;
    __shared__ float red[256];

    float v0 = outp[(size_t)row * 512 + t];
    float v1 = outp[(size_t)row * 512 + t + 256];

    red[t] = fmaxf(v0, v1);
    __syncthreads();
    for (int s = 128; s > 0; s >>= 1) {
        if (t < s) red[t] = fmaxf(red[t], red[t + s]);
        __syncthreads();
    }
    const float M = red[0];
    __syncthreads();

    float e0 = __expf(v0 - M), e1 = __expf(v1 - M);
    red[t] = e0 + e1;
    __syncthreads();
    for (int s = 128; s > 0; s >>= 1) {
        if (t < s) red[t] += red[t + s];
        __syncthreads();
    }
    const float S = red[0];

    out[(size_t)row * 512 + t]       = e0 / S;
    out[(size_t)row * 512 + t + 256] = e1 / S;
}

// ---------------------------------------------------------------------------
// Host launcher. Inputs: x(f32), edge_index(i64, 2xE), W(f32), a(f32).
// Workspace layout (~19.3 MB):
//   hb   : bf16 h, B-frag layout    @ 0         (4 MB)
//   xa   : bf16 x, A-frag layout    @ 4194304   (4 MB)
//   wf   : bf16 W, B-frag layout    @ 8388608   (512 KB)
//   ssrc : f32, H*N                 @ 8912896   (128 KB)
//   sdst : f32, H*N                 @ 9043968   (128 KB)
//   adj  : u32 bitmask, N*128       @ 9175040   (2 MB)
//   outp : f32, N*512               @ 11272192  (8 MB)
// ---------------------------------------------------------------------------
extern "C" void kernel_launch(void* const* d_in, const int* in_sizes, int n_in,
                              void* d_out, int out_size, void* d_ws, size_t ws_size,
                              hipStream_t stream) {
    const float*     x   = (const float*)d_in[0];
    const long long* ei  = (const long long*)d_in[1];
    const float*     W   = (const float*)d_in[2];
    const float*     a   = (const float*)d_in[3];
    float*           out = (float*)d_out;

    char* ws = (char*)d_ws;
    unsigned short* hb   = (unsigned short*)(ws);
    unsigned short* xa   = (unsigned short*)(ws + 4194304);
    unsigned short* wf   = (unsigned short*)(ws + 8388608);
    float*          ssrc = (float*)(ws + 8912896);
    float*          sdst = (float*)(ws + 9043968);
    unsigned*       adj  = (unsigned*)(ws + 9175040);
    float*          outp = (float*)(ws + 11272192);

    k_convx<<<NTILES * KTILES, 32, 0, stream>>>(x, xa);
    k_convw<<<GAT_H * KTILES * 4, 32, 0, stream>>>(W, wf);
    k_project<<<GAT_H * NTILES, 32, 0, stream>>>(xa, wf, a, hb, ssrc, sdst);
    k_adj_init<<<GAT_N, 128, 0, stream>>>(adj);
    k_adj_scatter<<<GAT_E / 256, 256, 0, stream>>>(ei, ei + GAT_E, adj);
    k_attn<<<GAT_H * NTILES, 32, 0, stream>>>(hb, ssrc, sdst, adj, outp);
    k_rowsoftmax<<<GAT_N, 256, 0, stream>>>(outp, out);
}
